// Pointnet2ClsMSGFus_raw2_79061757985085
// MI455X (gfx1250) — matrix operand order repaired and checked by reading, not compiled, not run on hardware
//
#include <hip/hip_runtime.h>
#include <math.h>

typedef __attribute__((ext_vector_type(16))) _Float16 v16h;
typedef __attribute__((ext_vector_type(8)))  _Float16 v8h;
typedef __attribute__((ext_vector_type(8)))  float    v8f;

// ------------------------------------------------------------------
// Split pointcloud [B,1024,387] -> xyz [B,1024,3], feats [B,384,1024]
// ------------------------------------------------------------------
__global__ void split_pc_kernel(const float* __restrict__ pc, float* __restrict__ xyz,
                                float* __restrict__ feats, int Npts, int Cin) {
  size_t t = (size_t)blockIdx.x * blockDim.x + threadIdx.x;
  size_t total = (size_t)8 * Npts * Cin;
  if (t >= total) return;
  int c = (int)(t % Cin);
  size_t bn = t / Cin;
  int nn = (int)(bn % Npts);
  int b  = (int)(bn / Npts);
  float v = pc[t];
  if (c < 3) xyz[((size_t)b * Npts + nn) * 3 + c] = v;
  else       feats[((size_t)b * 384 + (c - 3)) * Npts + nn] = v;
}

// ------------------------------------------------------------------
// Farthest point sampling, one block per batch. Npts in {256,512,1024}.
// ------------------------------------------------------------------
__global__ void fps_kernel(const float* __restrict__ xyz, int Npts, int npoint,
                           int* __restrict__ idx) {
  int b = blockIdx.x;
  int tid = threadIdx.x;
  const float* X = xyz + (size_t)b * Npts * 3;
  int cnt = Npts >> 8;            // points per thread (1,2,4)
  float px[4], py[4], pz[4], dmin[4];
  for (int q = 0; q < cnt; ++q) {
    int p = tid + (q << 8);
    px[q] = X[p * 3 + 0]; py[q] = X[p * 3 + 1]; pz[q] = X[p * 3 + 2];
    dmin[q] = 1e10f;
  }
  __shared__ float sv[256];
  __shared__ int   si[256];
  __shared__ float slast[3];
  __shared__ int   sbest;
  if (tid == 0) {
    idx[(size_t)b * npoint] = 0;
    slast[0] = X[0]; slast[1] = X[1]; slast[2] = X[2];
  }
  __syncthreads();
  for (int it = 1; it < npoint; ++it) {
    float lx = slast[0], ly = slast[1], lz = slast[2];
    float bv = -1.f; int bi = 0;
    for (int q = 0; q < cnt; ++q) {
      float dx = px[q] - lx, dy = py[q] - ly, dz = pz[q] - lz;
      float d = dx * dx + dy * dy + dz * dz;
      float dm = fminf(dmin[q], d);
      dmin[q] = dm;
      int p = tid + (q << 8);
      if (dm > bv || (dm == bv && p < bi)) { bv = dm; bi = p; }
    }
    sv[tid] = bv; si[tid] = bi;
    __syncthreads();
    for (int s = 128; s > 0; s >>= 1) {
      if (tid < s) {
        float ov = sv[tid + s]; int oi = si[tid + s];
        if (ov > sv[tid] || (ov == sv[tid] && oi < si[tid])) { sv[tid] = ov; si[tid] = oi; }
      }
      __syncthreads();
    }
    if (tid == 0) { sbest = si[0]; idx[(size_t)b * npoint + it] = si[0]; }
    __syncthreads();
    int best = sbest;
    if ((best & 255) == tid) {
      int q = best >> 8;
      slast[0] = px[q]; slast[1] = py[q]; slast[2] = pz[q];
    }
    __syncthreads();
  }
}

// ------------------------------------------------------------------
// Ball query: ascending-index neighbors within radius, pad with first
// ------------------------------------------------------------------
__global__ void ball_query_kernel(const float* __restrict__ xyz, const float* __restrict__ nxyz,
                                  int Npts, int S, float r2, int ns, int* __restrict__ nidx) {
  int t = blockIdx.x * blockDim.x + threadIdx.x;
  if (t >= 8 * S) return;
  int b = t / S, s = t % S;
  const float* q = nxyz + ((size_t)b * S + s) * 3;
  float qx = q[0], qy = q[1], qz = q[2];
  const float* X = xyz + (size_t)b * Npts * 3;
  int* out = nidx + ((size_t)b * S + s) * ns;
  int cnt = 0;
  for (int j = 0; j < Npts && cnt < ns; ++j) {
    float dx = X[j * 3 + 0] - qx, dy = X[j * 3 + 1] - qy, dz = X[j * 3 + 2] - qz;
    if (dx * dx + dy * dy + dz * dz < r2) out[cnt++] = j;
  }
  int first = (cnt > 0) ? out[0] : 0;
  for (int j = cnt; j < ns; ++j) out[j] = first;
}

// ------------------------------------------------------------------
// Gather kernels
// ------------------------------------------------------------------
__global__ void gather_xyz_kernel(const float* __restrict__ xyz, const int* __restrict__ idx,
                                  float* __restrict__ out, int Npts, int S) {
  int t = blockIdx.x * blockDim.x + threadIdx.x;
  if (t >= 8 * S * 3) return;
  int c = t % 3; int rest = t / 3; int s = rest % S; int b = rest / S;
  int p = idx[(size_t)b * S + s];
  out[((size_t)b * S + s) * 3 + c] = xyz[((size_t)b * Npts + p) * 3 + c];
}

__global__ void gather_feats_kernel(const float* __restrict__ fin, const int* __restrict__ idx,
                                    float* __restrict__ fout, int Npts, int S) {
  int t = blockIdx.x * blockDim.x + threadIdx.x;
  if (t >= 8 * 384 * S) return;
  int s = t % S; int rest = t / S; int c = rest % 384; int b = rest / 384;
  int p = idx[(size_t)b * S + s];
  fout[((size_t)b * 384 + c) * S + s] = fin[((size_t)b * 384 + c) * Npts + p];
}

// ------------------------------------------------------------------
// Pack fp32 weights [O,K] into A-fragment layout (f16), zero-padded.
// ------------------------------------------------------------------
__global__ void pack_weights_kernel(const float* __restrict__ W, _Float16* __restrict__ Wfrag,
                                    int O, int K, int Otiles, int nkblk) {
  int t = blockIdx.x * blockDim.x + threadIdx.x;
  int total = Otiles * nkblk * 32;
  if (t >= total) return;
  int lane = t & 31;
  int rest = t >> 5;
  int kblk = rest % nkblk;
  int otile = rest / nkblk;
  int o = otile * 16 + (lane & 15);
  int half = lane >> 4;
  v8h lo, hi;
  for (int j = 0; j < 8; ++j) {
    int k1 = kblk * 32 + half * 16 + j;
    int k2 = k1 + 8;
    lo[j] = (_Float16)((o < O && k1 < K) ? W[(size_t)o * K + k1] : 0.f);
    hi[j] = (_Float16)((o < O && k2 < K) ? W[(size_t)o * K + k2] : 0.f);
  }
  size_t off = (((size_t)otile * nkblk + kblk) * 32 + lane) * 16;
  *(v8h*)(Wfrag + off)     = lo;
  *(v8h*)(Wfrag + off + 8) = hi;
}

// ------------------------------------------------------------------
// Build grouped input directly in B-fragment layout (f16), one batch.
// ------------------------------------------------------------------
__global__ void group_frag_kernel(const float* __restrict__ xyz, const float* __restrict__ new_xyz,
                                  const int* __restrict__ nidx,
                                  const float* __restrict__ f1, int C1,
                                  const float* __restrict__ f2, int C2,
                                  _Float16* __restrict__ Xfrag,
                                  int b, int Npts, int S, int n, int Kreal, int nkblk, int Mtiles) {
  int t = blockIdx.x * blockDim.x + threadIdx.x;
  int total = Mtiles * nkblk * 32;
  if (t >= total) return;
  int lane = t & 31;
  int rest = t >> 5;
  int kblk = rest % nkblk;
  int mtile = rest / nkblk;
  int col = mtile * 16 + (lane & 15);
  int r0 = kblk * 32 + (lane >> 4) * 16;
  int s = col / n, i = col % n;
  int pt; float cx, cy, cz;
  if (nidx) {
    pt = nidx[((size_t)b * S + s) * n + i];
    const float* c = new_xyz + ((size_t)b * S + s) * 3;
    cx = c[0]; cy = c[1]; cz = c[2];
  } else {
    pt = col; cx = cy = cz = 0.f;   // sa_all: raw xyz channels, pt == column
  }
  const float* P = xyz + ((size_t)b * Npts + pt) * 3;
  v8h out[2];
  for (int j = 0; j < 16; ++j) {
    int k = r0 + j;
    float v;
    if (k < 3)            v = P[k] - (k == 0 ? cx : (k == 1 ? cy : cz));
    else if (k < 3 + C1)  v = f1[((size_t)b * C1 + (k - 3)) * Npts + pt];
    else if (k < Kreal)   v = f2[((size_t)b * C2 + (k - 3 - C1)) * Npts + pt];
    else                  v = 0.f;
    out[j >> 3][j & 7] = (_Float16)v;
  }
  size_t off = (((size_t)mtile * nkblk + kblk) * 32 + lane) * 16;
  *(v8h*)(Xfrag + off)     = out[0];
  *(v8h*)(Xfrag + off + 8) = out[1];
}

// ------------------------------------------------------------------
// Fused WMMA GEMM with 2x2 register blocking:
//   each wave computes a 32(O) x 32(M) block = 4 accumulators.
//   Per K-step: 2 A frags + 2 B frags (4 x b128x2) -> 4 v_wmma_f32_16x16x32_f16.
//   32 B/lane per WMMA vs 64 unblocked: 2x arithmetic intensity.
// Output stored in next-layer B-fragment layout; padded rows forced to 0.
// ------------------------------------------------------------------
__global__ void __launch_bounds__(128)
wmma_mlp2x2_kernel(const _Float16* __restrict__ Xfrag, const _Float16* __restrict__ Wfrag,
                   const float* __restrict__ gamma, const float* __restrict__ beta,
                   _Float16* __restrict__ Yfrag,
                   int Mt2, int Ot2, int nkblk, int nkblkO, int O) {
  int wave = threadIdx.x >> 5;
  int lane = threadIdx.x & 31;
  int tile = blockIdx.x * 4 + wave;
  if (tile >= Mt2 * Ot2) return;          // wave-uniform exit: EXEC stays all-ones
  int mtile0 = (tile % Mt2) * 2;
  int otile0 = (tile / Mt2) * 2;

  size_t ostride = (size_t)nkblk * 64;    // v8h units per fragment stream
  const v8h* __restrict__ A8 = (const v8h*)Wfrag + (size_t)otile0 * ostride + lane * 2;
  const v8h* __restrict__ B8 = (const v8h*)Xfrag + (size_t)mtile0 * ostride + lane * 2;

  v8f acc00 = {}, acc01 = {}, acc10 = {}, acc11 = {};
  union U16 { v16h v; v8h h[2]; };
  for (int kb = 0; kb < nkblk; ++kb) {
    U16 a0, a1, b0, b1;
    a0.h[0] = A8[0];           a0.h[1] = A8[1];
    a1.h[0] = A8[ostride];     a1.h[1] = A8[ostride + 1];
    b0.h[0] = B8[0];           b0.h[1] = B8[1];
    b1.h[0] = B8[ostride];     b1.h[1] = B8[ostride + 1];
    A8 += 64; B8 += 64;
    __builtin_prefetch((const void*)A8, 0, 1);   // global_prefetch_b8
    __builtin_prefetch((const void*)B8, 0, 1);
    acc00 = __builtin_amdgcn_wmma_f32_16x16x32_f16(false, a0.v, false, b0.v, (short)0, acc00, false, false);
    acc01 = __builtin_amdgcn_wmma_f32_16x16x32_f16(false, a0.v, false, b1.v, (short)0, acc01, false, false);
    acc10 = __builtin_amdgcn_wmma_f32_16x16x32_f16(false, a1.v, false, b0.v, (short)0, acc10, false, false);
    acc11 = __builtin_amdgcn_wmma_f32_16x16x32_f16(false, a1.v, false, b1.v, (short)0, acc11, false, false);
  }

  int m  = lane & 15;
  int hi = lane >> 4;
  v8f* accs[4] = {&acc00, &acc01, &acc10, &acc11};
  for (int oi = 0; oi < 2; ++oi) {
    int otile = otile0 + oi;
    int obase = otile * 16 + hi * 8;        // acc[i] -> row obase+i (ISA C/D layout)
    for (int mi = 0; mi < 2; ++mi) {
      int mtile = mtile0 + mi;
      v8f acc = *accs[oi * 2 + mi];
      v8h ov;
      for (int i = 0; i < 8; ++i) {
        int o = obase + i;
        float g  = (o < O) ? gamma[o] : 0.f;
        float bb = (o < O) ? beta[o]  : 0.f;
        float v = g * acc[i] + bb;
        ov[i] = (_Float16)(v > 0.f ? v : 0.f);
      }
      // next-layer B-fragment: kblk = otile>>1, lane' = (otile&1)*16 + m, j0 = hi*8
      size_t doff = ((((size_t)mtile * nkblkO) + (otile >> 1)) * 32 + ((otile & 1) * 16 + m)) * 16 + hi * 8;
      *(v8h*)(Yfrag + doff) = ov;
    }
  }
}

// ------------------------------------------------------------------
// Max-pool over neighbor dim from fragment layout -> li[b][coff+o][s] (f32)
// ------------------------------------------------------------------
__global__ void pool_max_kernel(const _Float16* __restrict__ Yfrag, float* __restrict__ li,
                                int O, int nkblkO, int n, int S, int ldC, int coff, int b) {
  int t = blockIdx.x * blockDim.x + threadIdx.x;
  if (t >= O * S) return;
  int o = t / S, s = t % S;
  int kblk = o >> 5;
  int lphalf = (o >> 4) & 1;
  int j = o & 15;
  float mx = -3.4e38f;
  for (int i = 0; i < n; ++i) {
    int m = s * n + i;
    size_t off = (((size_t)(m >> 4) * nkblkO + kblk) * 32 + (lphalf * 16 + (m & 15))) * 16 + j;
    mx = fmaxf(mx, (float)Yfrag[off]);
  }
  li[((size_t)b * ldC + coff + o) * S + s] = mx;
}

// ------------------------------------------------------------------
// Point attention: h = relu(g*W1@li+b); a = sigmoid(W2@h+bias); out = li*a
// ------------------------------------------------------------------
__global__ void attn_h_kernel(const float* __restrict__ li, const float* __restrict__ W1,
                              const float* __restrict__ g, const float* __restrict__ bb,
                              float* __restrict__ h, int C, int hid, int S) {
  int t = blockIdx.x * blockDim.x + threadIdx.x;
  if (t >= 8 * hid * S) return;
  int s = t % S; int rest = t / S; int o = rest % hid; int b = rest / hid;
  const float* w = W1 + (size_t)o * C;
  const float* x = li + (size_t)b * C * S + s;
  float sum = 0.f;
  for (int c = 0; c < C; ++c) sum += w[c] * x[(size_t)c * S];
  float v = g[o] * sum + bb[o];
  h[((size_t)b * hid + o) * S + s] = v > 0.f ? v : 0.f;
}

__global__ void attn_a_kernel(const float* __restrict__ h, const float* __restrict__ W2,
                              const float* __restrict__ bias, float* __restrict__ a,
                              int hid, int S) {
  int t = blockIdx.x * blockDim.x + threadIdx.x;
  if (t >= 8 * S) return;
  int s = t % S; int b = t / S;
  float sum = 0.f;
  for (int o = 0; o < hid; ++o) sum += W2[o] * h[((size_t)b * hid + o) * S + s];
  sum += bias[0];
  a[(size_t)b * S + s] = 1.f / (1.f + __expf(-sum));
}

__global__ void attn_apply_kernel(const float* __restrict__ li, const float* __restrict__ a,
                                  float* __restrict__ out, int C, int S) {
  int t = blockIdx.x * blockDim.x + threadIdx.x;
  if (t >= 8 * C * S) return;
  int s = t % S; int rest = t / S; int b = rest / C;
  out[t] = li[t] * a[(size_t)b * S + s];
}

// ==================================================================
// Host orchestration
// ==================================================================
extern "C" void kernel_launch(void* const* d_in, const int* in_sizes, int n_in,
                              void* d_out, int out_size, void* d_ws, size_t ws_size,
                              hipStream_t stream) {
  (void)in_sizes; (void)n_in; (void)out_size; (void)ws_size;
  const int B = 8;
  const float* pc = (const float*)d_in[0];
  auto saP = [&](int k, int br, int ly, int t) -> const float* {
    return (const float*)d_in[1 + ((k * 2 + br) * 3 + ly) * 3 + t];
  };
  auto atP = [&](int k, int t) -> const float* {
    return (const float*)d_in[73 + k * 5 + t];
  };

  char* wsb = (char*)d_ws; size_t off = 0;
  auto alloc = [&](size_t bytes) -> char* {
    off = (off + 255) & ~(size_t)255;
    char* p = wsb + off; off += bytes; return p;
  };

  float* xyz0 = (float*)alloc((size_t)B * 1024 * 3 * 4);
  float* xyz1 = (float*)alloc((size_t)B * 512 * 3 * 4);
  float* xyz2 = (float*)alloc((size_t)B * 256 * 3 * 4);
  float* xyz3 = (float*)alloc((size_t)B * 128 * 3 * 4);
  float* xyzs[4] = {xyz0, xyz1, xyz2, xyz3};
  float* featsA = (float*)alloc((size_t)B * 384 * 1024 * 4);
  float* featsB = (float*)alloc((size_t)B * 384 * 1024 * 4);
  int*   fpsidx = (int*)alloc((size_t)B * 512 * 4);
  int*   nidx   = (int*)alloc((size_t)B * 512 * 64 * 4);
  float* li     = (float*)alloc((size_t)B * 65536 * 4);
  float* curf   = (float*)alloc((size_t)B * 65536 * 4);
  float* hbuf   = (float*)alloc((size_t)B * 16384 * 4);
  float* abuf   = (float*)alloc((size_t)B * 512 * 4);
  _Float16* wfrags[3];
  wfrags[0] = (_Float16*)alloc(1u << 20);
  wfrags[1] = (_Float16*)alloc(1u << 20);
  wfrags[2] = (_Float16*)alloc(1u << 20);
  _Float16* xfrag  = (_Float16*)alloc((size_t)28 * 1024 * 1024);
  _Float16* yfragA = (_Float16*)alloc((size_t)4 * 1024 * 1024 + 65536);
  _Float16* yfragB = (_Float16*)alloc((size_t)4 * 1024 * 1024 + 65536);

  {
    size_t total = (size_t)B * 1024 * 387;
    split_pc_kernel<<<(unsigned)((total + 255) / 256), 256, 0, stream>>>(pc, xyz0, featsA, 1024, 387);
  }

  const int NIN[4]  = {1024, 512, 256, 128};
  const int SS[4]   = {512, 256, 128, 1};
  const int NSAMP[4][2] = {{32, 64}, {16, 32}, {8, 16}, {128, 128}};
  const float RAD2[4][2] = {{0.0004f, 0.0016f}, {0.0016f, 0.0064f}, {0.0064f, 0.0256f}, {0.f, 0.f}};
  const int SPEC[4][2][3] = {{{16, 16, 32}, {32, 32, 64}},
                             {{64, 64, 128}, {64, 96, 128}},
                             {{128, 196, 256}, {128, 196, 256}},
                             {{256, 256, 512}, {256, 384, 512}}};
  const int CHOUT[4] = {96, 256, 512, 1024};
  const int HID[4]   = {24, 64, 128, 256};

  float* featsCur = featsA;
  float* featsNext = featsB;

  for (int k = 0; k < 4; ++k) {
    int Npts = NIN[k], S = SS[k];
    float* curxyz = xyzs[k];
    if (k < 3) {
      fps_kernel<<<B, 256, 0, stream>>>(curxyz, Npts, S, fpsidx);
      size_t tg = (size_t)B * S * 3;
      gather_xyz_kernel<<<(unsigned)((tg + 255) / 256), 256, 0, stream>>>(curxyz, fpsidx, xyzs[k + 1], Npts, S);
    }
    int C1 = (k == 0) ? 384 : CHOUT[k - 1];
    int C2 = (k == 0) ? 0 : 384;
    const float* f1 = (k == 0) ? (const float*)featsCur : (const float*)curf;
    const float* f2 = (k == 0) ? (const float*)nullptr : (const float*)featsCur;
    int Kreal = 3 + C1 + C2;
    int nkblkIn = (Kreal + 31) >> 5;

    for (int br = 0; br < 2; ++br) {
      int n = NSAMP[k][br], M = S * n, Mtiles = M >> 4;
      if (k < 3) {
        size_t tq = (size_t)B * S;
        ball_query_kernel<<<(unsigned)((tq + 255) / 256), 256, 0, stream>>>(
            curxyz, xyzs[k + 1], Npts, S, RAD2[k][br], n, nidx);
      }
      int dims[4] = {Kreal, SPEC[k][br][0], SPEC[k][br][1], SPEC[k][br][2]};
      for (int ly = 0; ly < 3; ++ly) {
        int K = dims[ly], O = dims[ly + 1];
        int Op32 = (O + 31) & ~31;
        int Otiles = Op32 >> 4;
        int nkb = (K + 31) >> 5;
        int tot = Otiles * nkb * 32;
        pack_weights_kernel<<<(tot + 255) / 256, 256, 0, stream>>>(saP(k, br, ly, 0), wfrags[ly], O, K, Otiles, nkb);
      }
      int coff = (br == 0) ? 0 : SPEC[k][0][2];
      for (int b = 0; b < B; ++b) {
        {
          int tot = Mtiles * nkblkIn * 32;
          group_frag_kernel<<<(tot + 255) / 256, 256, 0, stream>>>(
              curxyz, (k < 3) ? xyzs[k + 1] : nullptr, (k < 3) ? nidx : nullptr,
              f1, C1, f2, C2, xfrag, b, Npts, S, n, Kreal, nkblkIn, Mtiles);
        }
        const _Float16* inF = xfrag;
        int inkb = nkblkIn;
        int OlastP32 = 32;
        for (int ly = 0; ly < 3; ++ly) {
          int O = dims[ly + 1];
          int Op32 = (O + 31) & ~31;
          int Otiles = Op32 >> 4;
          _Float16* outF = (ly & 1) ? yfragB : yfragA;
          int Mt2 = Mtiles >> 1, Ot2 = Otiles >> 1;   // all tile counts are even
          int tiles2 = Mt2 * Ot2;
          wmma_mlp2x2_kernel<<<(tiles2 + 3) / 4, 128, 0, stream>>>(
              inF, wfrags[ly], saP(k, br, ly, 1), saP(k, br, ly, 2),
              outF, Mt2, Ot2, inkb, Op32 >> 5, O);
          inF = outF; inkb = Op32 >> 5; OlastP32 = Op32;
        }
        int Olast = dims[3];
        int tot = Olast * S;
        pool_max_kernel<<<(tot + 255) / 256, 256, 0, stream>>>(
            inF, li, Olast, OlastP32 >> 5, n, S, CHOUT[k], coff, b);
      }
    }
    // attention + apply
    {
      int C = CHOUT[k], hid = HID[k];
      size_t th = (size_t)B * hid * S;
      attn_h_kernel<<<(unsigned)((th + 255) / 256), 256, 0, stream>>>(
          li, atP(k, 0), atP(k, 1), atP(k, 2), hbuf, C, hid, S);
      size_t ta = (size_t)B * S;
      attn_a_kernel<<<(unsigned)((ta + 255) / 256), 256, 0, stream>>>(
          hbuf, atP(k, 3), atP(k, 4), abuf, hid, S);
      float* dst = (k == 3) ? (float*)d_out : curf;
      size_t tm = (size_t)B * C * S;
      attn_apply_kernel<<<(unsigned)((tm + 255) / 256), 256, 0, stream>>>(li, abuf, dst, C, S);
    }
    if (k < 3) {
      size_t tf = (size_t)B * 384 * S;
      gather_feats_kernel<<<(unsigned)((tf + 255) / 256), 256, 0, stream>>>(featsCur, fpsidx, featsNext, Npts, S);
      float* tmp = featsCur; featsCur = featsNext; featsNext = tmp;
    }
  }
}
